// Attention_1683627180455
// MI455X (gfx1250) — compile-verified, matrix operand order checked by reading, not run on hardware
//
#include <hip/hip_runtime.h>

// ---------------------------------------------------------------------------
// Problem constants (from reference)
// ---------------------------------------------------------------------------
#define DMODEL 512
#define NHEAD  8
#define DHEAD  64          // DMODEL / NHEAD
#define BATCH  4
#define SEQ    512
#define TGT    32
#define BS     (BATCH * SEQ)   // 2048 query rows
#define MKV    (BS * TGT)      // 65536 kv rows

#define KSTEPS (DMODEL / 32)   // 16 WMMA K-steps (compile-time)
#define NITER  (KSTEPS * 8)    // 128 (kk,sub) WMMAs per wave
#define LDSK   (DMODEL + 8)    // padded LDS row stride (halves): 1040B = 4 dwords
                               // mod 64 banks -> conflict-free fragment reads
#define LDS_BYTES (128 * LDSK * 2)

typedef __bf16 bf16_t;
typedef __attribute__((ext_vector_type(16))) __bf16 v16bf;
typedef __attribute__((ext_vector_type(8)))  __bf16 v8bf;
typedef __attribute__((ext_vector_type(8)))  float  v8f;

// ---------------------------------------------------------------------------
// 16-bit A-matrix fragment layout (ISA 7.12.2): lane l holds row m = l&15.
//   lanes 0-15 : halves 0..7 = K k0+0..7,   halves 8..15 = K k0+16..23
//   lanes 16-31: halves 0..7 = K k0+8..15,  halves 8..15 = K k0+24..31
// => two contiguous 16B runs at p and p+16, lane-half K-base folded into p.
// B fragments are loaded symmetrically (lane = output column).
// ---------------------------------------------------------------------------

// Raw (pre-conversion) A-fragment staging, so loads can be issued a K-step
// ahead of the f32->bf16 converts that consume them.
template <typename AT> struct ARaw;

template <> struct ARaw<float> {
  v8f lo, hi;
  __device__ __forceinline__ void load(const float* p) {
    lo = *(const v8f*)(p);
    hi = *(const v8f*)(p + 16);
  }
  __device__ __forceinline__ v16bf frag() const {
    v16bf r;
#pragma unroll
    for (int i = 0; i < 8; ++i) { r[i] = (__bf16)lo[i]; r[i + 8] = (__bf16)hi[i]; }
    return r;
  }
};

template <> struct ARaw<bf16_t> {
  v8bf lo, hi;
  __device__ __forceinline__ void load(const bf16_t* p) {
    lo = *(const v8bf*)(p);
    hi = *(const v8bf*)(p + 16);
  }
  __device__ __forceinline__ v16bf frag() const {
    return __builtin_shufflevector(lo, hi, 0, 1, 2, 3, 4, 5, 6, 7,
                                   8, 9, 10, 11, 12, 13, 14, 15);
  }
};

__device__ __forceinline__ v16bf lds_frag(const bf16_t* p) {
  v8bf lo = *(const v8bf*)(p);
  v8bf hi = *(const v8bf*)(p + 16);
  return __builtin_shufflevector(lo, hi, 0, 1, 2, 3, 4, 5, 6, 7,
                                 8, 9, 10, 11, 12, 13, 14, 15);
}

// ---------------------------------------------------------------------------
// Tiled WMMA GEMM:  C[M,N] = A[M,K=512] * W[N,K]^T + bias[N]
//   block = 256 threads = 8 waves; block tile = 128(M) x 128(N)
//   W tile (128 x 512 bf16, 130KB padded) staged once in LDS
//   wave tile = 16(M) x 128(N): one A fragment feeds 8 v_wmma
//   Two-level nest (16 x 8) -- both trip counts small enough that
//   #pragma unroll reliably produces straight-line SSA code:
//     - depth-1 B pipeline with two NAMED v16bf values (no arrays/modulo, so
//       no risk of dynamic VGPR indexing): each WMMA consumes the fragment
//       whose ds_loads were issued one WMMA earlier
//     - A raw loads ping-pong one K-step ahead (first load primed before the
//       staging barrier so it hides under the W-tile copy)
//     - two LDS base pointers so all ds offsets fit the 16-bit offset field
//   FINAL=false : store bf16 into Cbf
//   FINAL=true  : store f32 into Cf, zeroing rows flagged in rowmask
// ---------------------------------------------------------------------------
template <bool FINAL, typename AT>
__global__ __launch_bounds__(256) void gemm_wmma_kernel(
    const AT* __restrict__ A, const bf16_t* __restrict__ W,
    const float* __restrict__ bias,
    bf16_t* __restrict__ Cbf, float* __restrict__ Cf,
    const unsigned char* __restrict__ rowmask,
    int M, int N)
{
  extern __shared__ bf16_t Wl[];     // [128][LDSK]

  const int tid  = threadIdx.x;
  const int lane = tid & 31;
  const int wid  = tid >> 5;                 // 0..7 : M sub-tile
  const int m0   = blockIdx.y * 128 + wid * 16;
  const int n0   = blockIdx.x * 128;
  const int l15  = lane & 15;
  const int kb   = (lane >> 4) << 3;         // 0 or 8 : lane-half K base

  const AT* arow = A + (size_t)(m0 + l15) * DMODEL + kb;

  // Prime A pipeline (kk = 0) before the staging barrier: this global load
  // completes under the W-tile copy.
  ARaw<AT> abuf[2];
  abuf[0].load(arow);

  // ---- stage W tile [128][512] -> LDS (16B chunks, all 256 threads) ----
  // chunksPerRow = 512/8 = 64
  for (int i = tid; i < 128 * 64; i += 256) {
    const int row = i >> 6;
    const int c   = (i & 63) << 3;
    *(v8bf*)(Wl + row * LDSK + c) =
        *(const v8bf*)(W + (size_t)(n0 + row) * DMODEL + c);
  }
  __syncthreads();

  // Two bases so every fragment's byte offset stays < 64KB (DS offset field).
  const bf16_t* bLo = Wl + (size_t)l15 * LDSK + kb;
  const bf16_t* bHi = bLo + (size_t)4 * 16 * LDSK;

  v8f acc[8] = {};

  // depth-1 B pipeline: bnxt always holds the next WMMA's fragment
  v16bf bnxt = lds_frag(bLo);                // (kk=0, sub=0)

#pragma unroll
  for (int kk = 0; kk < KSTEPS; ++kk) {
    const int k0 = kk * 32;
    if (kk + 1 < KSTEPS) {
      abuf[(kk + 1) & 1].load(arow + k0 + 32);     // next K-step's A (no wait)
      __builtin_prefetch(arow + k0 + 96, 0, 3);    // 2 steps ahead, WGP-near
    }
    const v16bf a = abuf[kk & 1].frag();
#pragma unroll
    for (int sub = 0; sub < 8; ++sub) {
      const v16bf bcur = bnxt;
      const int nit = kk * 8 + sub + 1;            // next (kk,sub), flattened
      if (nit < NITER) {
        const int nkk  = nit >> 3;
        const int nsub = nit & 7;
        const bf16_t* base = (nsub < 4) ? bLo : bHi;
        bnxt = lds_frag(base + (nsub & 3) * 16 * LDSK + nkk * 32);
      }
      acc[sub] = __builtin_amdgcn_wmma_f32_16x16x32_bf16(
          false, a, false, bcur, (short)0, acc[sub], false, false);
    }
  }

  // C/D layout: lanes 0-15 -> N=lane, M=r ; lanes 16-31 -> N=lane-16, M=8+r
  const int mh = (lane >> 4) << 3;
#pragma unroll
  for (int sub = 0; sub < 8; ++sub) {
    const int n = n0 + sub * 16 + l15;
    const float bn = bias[n];
#pragma unroll
    for (int r = 0; r < 8; ++r) {
      const int m = m0 + r + mh;
      float v = acc[sub][r] + bn;
      if (FINAL) {
        if (rowmask[m]) v = 0.0f;                // mask_no_tgt_valid rows -> 0
        Cf[(size_t)m * N + n] = v;
      } else {
        Cbf[(size_t)m * N + n] = (bf16_t)v;
      }
    }
  }
}

// ---------------------------------------------------------------------------
// Attention core. 1 block per (b,s); wave == head (wave32 maps exactly onto
// the 8x32 head/target grid); lane == target index.
//   scores -> scale 1/sqrt(64) -> mask (-inf unless whole row masked, matching
//   the reference's exemption) -> shfl softmax -> P@V via LDS-staged
//   probabilities (lane covers output dims j and j+32).
// kv rows are length 2*DMODEL: [0,512) = K, [512,1024) = V.
// ---------------------------------------------------------------------------
__global__ __launch_bounds__(256) void attn_kernel(
    const bf16_t* __restrict__ q, const bf16_t* __restrict__ kv,
    const unsigned char* __restrict__ mask,
    const unsigned char* __restrict__ rowmask,
    bf16_t* __restrict__ attnout)
{
  const int bs = blockIdx.x;
  const int h  = threadIdx.x >> 5;
  const int j  = threadIdx.x & 31;

  const bf16_t* qh   = q  + (size_t)bs * DMODEL + h * DHEAD;
  const bf16_t* krow = kv + ((size_t)bs * TGT + j) * (2 * DMODEL) + h * DHEAD;

  float s = 0.0f;
#pragma unroll 8
  for (int d = 0; d < DHEAD; ++d) s += (float)qh[d] * (float)krow[d];
  s *= 0.125f;                                   // 1/sqrt(DHEAD)

  const bool rall = rowmask[bs] != 0;
  if (mask[(size_t)bs * TGT + j] && !rall) s = -3.0e38f;

  float mx = s;
#pragma unroll
  for (int o = 16; o > 0; o >>= 1) mx = fmaxf(mx, __shfl_xor(mx, o, 32));
  float e = __expf(s - mx);
  float sum = e;
#pragma unroll
  for (int o = 16; o > 0; o >>= 1) sum += __shfl_xor(sum, o, 32);
  const float p = e / sum;

  __shared__ float lds_p[NHEAD][TGT];
  lds_p[h][j] = p;
  __syncthreads();

  const bf16_t* vbase = kv + (size_t)bs * TGT * (2 * DMODEL) + DMODEL + h * DHEAD;
  float o0 = 0.0f, o1 = 0.0f;
#pragma unroll 4
  for (int jj = 0; jj < TGT; ++jj) {
    const float pj = lds_p[h][jj];
    o0 += pj * (float)vbase[(size_t)jj * (2 * DMODEL) + j];
    o1 += pj * (float)vbase[(size_t)jj * (2 * DMODEL) + j + 32];
  }
  bf16_t* orow = attnout + (size_t)bs * DMODEL + h * DHEAD;
  orow[j]      = (bf16_t)o0;
  orow[j + 32] = (bf16_t)o1;
}

// ---------------------------------------------------------------------------
// Small helpers
// ---------------------------------------------------------------------------
__global__ void cvt_f32_bf16_kernel(const float* __restrict__ in,
                                    bf16_t* __restrict__ out, long long n)
{
  long long i = (long long)blockIdx.x * blockDim.x + threadIdx.x;
  if (i < n) out[i] = (bf16_t)in[i];
}

__global__ void rowmask_kernel(const unsigned char* __restrict__ mask,
                               unsigned char* __restrict__ rowmask, int rows)
{
  int r = blockIdx.x * blockDim.x + threadIdx.x;
  if (r < rows) {
    unsigned all = 1;
#pragma unroll
    for (int j = 0; j < TGT; ++j) all &= (unsigned)(mask[(size_t)r * TGT + j] != 0);
    rowmask[r] = (unsigned char)all;
  }
}

// ---------------------------------------------------------------------------
// Launch
// inputs: 0:src 1:tgt 2:tgt_padding_mask(bool,1B) 3:in_proj_weight
//         4:in_proj_bias 5:out_proj_weight 6:out_proj_bias
// ---------------------------------------------------------------------------
extern "C" void kernel_launch(void* const* d_in, const int* in_sizes, int n_in,
                              void* d_out, int out_size, void* d_ws, size_t ws_size,
                              hipStream_t stream)
{
  const float* src          = (const float*)d_in[0];
  const float* tgt          = (const float*)d_in[1];
  const unsigned char* mask = (const unsigned char*)d_in[2]; // jnp.bool_ = 1 byte
  const float* w_in         = (const float*)d_in[3];
  const float* b_in         = (const float*)d_in[4];
  const float* w_out        = (const float*)d_in[5];
  const float* b_out        = (const float*)d_in[6];
  float* out                = (float*)d_out;

  // Workspace layout (all offsets 256B-aligned); ~140 MB total.
  char* ws = (char*)d_ws;
  size_t off = 0;
  bf16_t* kv_bf   = (bf16_t*)(ws + off); off += (size_t)MKV * 2 * DMODEL * 2;   // 134 MB
  bf16_t* win_bf  = (bf16_t*)(ws + off); off += (size_t)3 * DMODEL * DMODEL * 2;
  bf16_t* wout_bf = (bf16_t*)(ws + off); off += (size_t)DMODEL * DMODEL * 2;
  bf16_t* q_bf    = (bf16_t*)(ws + off); off += (size_t)BS * DMODEL * 2;
  bf16_t* ao_bf   = (bf16_t*)(ws + off); off += (size_t)BS * DMODEL * 2;
  unsigned char* rmask = (unsigned char*)(ws + off); off += (size_t)BS;
  (void)ws_size; (void)in_sizes; (void)n_in; (void)out_size;

  // 1) weights -> bf16 (2 MB total; activations converted on the fly)
  {
    long long n = (long long)3 * DMODEL * DMODEL;
    cvt_f32_bf16_kernel<<<(unsigned)((n + 255) / 256), 256, 0, stream>>>(w_in, win_bf, n);
  }
  {
    long long n = (long long)DMODEL * DMODEL;
    cvt_f32_bf16_kernel<<<(unsigned)((n + 255) / 256), 256, 0, stream>>>(w_out, wout_bf, n);
  }

  // 2) per-(b,s) "all targets masked" flag
  rowmask_kernel<<<(BS + 255) / 256, 256, 0, stream>>>(mask, rmask, BS);

  // 3) Q projection: (2048 x 512) = src (2048 x 512) @ w_src^T
  gemm_wmma_kernel<false, float>
      <<<dim3(DMODEL / 128, BS / 128), 256, LDS_BYTES, stream>>>(
          src, win_bf, b_in, q_bf, nullptr, nullptr, BS, DMODEL);

  // 4) KV projection: (65536 x 1024) = tgt (65536 x 512) @ w_tgt^T (dominant)
  gemm_wmma_kernel<false, float>
      <<<dim3((2 * DMODEL) / 128, MKV / 128), 256, LDS_BYTES, stream>>>(
          tgt, win_bf + (size_t)DMODEL * DMODEL, b_in + DMODEL,
          kv_bf, nullptr, nullptr, MKV, 2 * DMODEL);

  // 5) masked softmax attention + P@V
  attn_kernel<<<BS, 256, 0, stream>>>(q_bf, kv_bf, mask, rmask, ao_bf);

  // 6) output projection + bias + zero fully-masked rows (f32 out)
  gemm_wmma_kernel<true, bf16_t>
      <<<dim3(DMODEL / 128, BS / 128), 256, LDS_BYTES, stream>>>(
          ao_bf, wout_bf, b_out, nullptr, out, rmask, BS, DMODEL);
}
